// TopKSAE_26860725469376
// MI455X (gfx1250) — matrix operand order repaired and checked by reading, not compiled, not run on hardware
//
#include <hip/hip_runtime.h>

// ---------------------------------------------------------------------------
// TopK SAE forward:  z = x @ W_enc^T + b_enc ; topk mask ; x_hat = z_s @ W_dec^T + b_dec
// B=8192, D_IN=768, D_HID=16384, K=64
// ---------------------------------------------------------------------------
constexpr int BB    = 8192;
constexpr int D_IN  = 768;
constexpr int D_HID = 16384;
constexpr int KK    = 64;

typedef __attribute__((ext_vector_type(16))) __bf16 v16bf;
typedef __attribute__((ext_vector_type(8)))  float  v8f;

union PairCvt { unsigned u; __bf16 h[2]; };

// --- CDNA5 async global->LDS path (guarded; falls back to sync staging) -----
// Only enabled in the device pass so the host-side parse never sees the
// aux-target builtin's address-space-qualified prototype.
#if defined(__AMDGCN__) && defined(__has_builtin)
#if __has_builtin(__builtin_amdgcn_global_load_async_to_lds_b128) && \
    __has_builtin(__builtin_amdgcn_s_wait_asynccnt)
#define HAVE_ASYNC_LDS 1
#endif
#endif
#ifndef HAVE_ASYNC_LDS
#define HAVE_ASYNC_LDS 0
#endif

#if HAVE_ASYNC_LDS
typedef int v4i_gcc __attribute__((vector_size(16)));                 // matches builtin pointee
typedef __attribute__((address_space(1))) v4i_gcc* as1_v4i_p;         // global (AS1)
typedef __attribute__((address_space(3))) v4i_gcc* as3_v4i_p;         // LDS (AS3)
#endif

__device__ __forceinline__ void stage16(const void* g, void* l) {
#if HAVE_ASYNC_LDS
  // inttoptr casts: generic LDS address carries the LDS offset in low 32 bits
  as1_v4i_p gp = (as1_v4i_p)(unsigned long long)g;
  as3_v4i_p lp = (as3_v4i_p)(unsigned int)(unsigned long long)l;
  __builtin_amdgcn_global_load_async_to_lds_b128(gp, lp, 0, 0);
#else
  *(uint4*)l = *(const uint4*)g;
#endif
}

__device__ __forceinline__ void stage_wait_3() {
#if HAVE_ASYNC_LDS
  __builtin_amdgcn_s_wait_asynccnt(3);
#endif
}
__device__ __forceinline__ void stage_wait_0() {
#if HAVE_ASYNC_LDS
  __builtin_amdgcn_s_wait_asynccnt(0);
#endif
}

// ---------------------------------------------------------------------------
// Kernel 0: f32 -> bf16 conversion of x and W_enc (one pass, grid-stride)
// ---------------------------------------------------------------------------
__global__ __launch_bounds__(256)
void cvt_kernel(const float* __restrict__ x, const float* __restrict__ we,
                __bf16* __restrict__ xb, __bf16* __restrict__ wb) {
  const size_t n1 = (size_t)BB * D_IN;
  const size_t n2 = (size_t)D_HID * D_IN;
  const size_t stride = (size_t)gridDim.x * blockDim.x;
  for (size_t i = (size_t)blockIdx.x * blockDim.x + threadIdx.x; i < n1 + n2; i += stride) {
    if (i < n1) xb[i] = (__bf16)x[i];
    else        wb[i - n1] = (__bf16)we[i - n1];
  }
}

// ---------------------------------------------------------------------------
// Kernel 1: transpose W_dec [D_IN, D_HID] -> W_decT [D_HID, D_IN]
// ---------------------------------------------------------------------------
__global__ __launch_bounds__(256)
void transpose_kernel(const float* __restrict__ wd, float* __restrict__ wdT) {
  __shared__ float tile[32][33];
  const int h0 = blockIdx.x * 32;
  const int d0 = blockIdx.y * 32;
  const int tx = threadIdx.x;      // 0..31
  const int ty = threadIdx.y;      // 0..7
#pragma unroll
  for (int i = 0; i < 32; i += 8)
    tile[ty + i][tx] = wd[(size_t)(d0 + ty + i) * D_HID + h0 + tx];
  __syncthreads();
#pragma unroll
  for (int i = 0; i < 32; i += 8)
    wdT[(size_t)(h0 + ty + i) * D_IN + d0 + tx] = tile[tx][ty + i];
}

// ---------------------------------------------------------------------------
// Kernel 2: encode GEMM with WMMA bf16 -> f32, double-buffered LDS staging.
//   Block tile: 64 (M) x 128 (N); 8 waves, each 32x32 via 2x2 of 16x16x32.
//   K loop over D_IN in steps of 32; stage k+1 streams (async if available)
//   into the alternate LDS buffer while WMMAs consume stage k.
// ---------------------------------------------------------------------------
__global__ __launch_bounds__(256)
void encode_kernel(const __bf16* __restrict__ xb, const __bf16* __restrict__ wb,
                   const float* __restrict__ b_enc, float* __restrict__ z) {
  constexpr int LDA = 40;                          // padded LDS row stride (elems)
  constexpr int NK  = D_IN / 32;                   // 24 K-steps
  __shared__ unsigned short As[2][64 * LDA];       // 2 x (64x32 bf16 A tile)
  __shared__ unsigned short Bs[2][128 * LDA];      // 2 x (128x32 bf16 B tile)

  const int tid  = threadIdx.x;
  const int lane = tid & 31;
  const int wave = tid >> 5;
  const int wm   = wave & 1;                       // M sub-tile (0..1)
  const int wn   = wave >> 1;                      // N sub-tile (0..3)
  const int m0   = blockIdx.y * 64;
  const int n0   = blockIdx.x * 128;

  const int ldr = tid >> 2;                        // 0..63
  const int ldc = (tid & 3) * 8;                   // 0,8,16,24 (bf16 elems)

  const __bf16* gA  = xb + (size_t)(m0 + ldr) * D_IN + ldc;
  const __bf16* gB0 = wb + (size_t)(n0 + ldr) * D_IN + ldc;
  const __bf16* gB1 = wb + (size_t)(n0 + 64 + ldr) * D_IN + ldc;

  v8f acc[2][2];
#pragma unroll
  for (int mi = 0; mi < 2; ++mi)
#pragma unroll
    for (int ni = 0; ni < 2; ++ni)
#pragma unroll
      for (int e = 0; e < 8; ++e) acc[mi][ni][e] = 0.0f;

  const int l15 = lane & 15;
  const int ab  = (lane & 16) ? 8  : 0;            // A-frag K base (ISA 16-bit A layout)
  const int bbo = (lane & 16) ? 16 : 0;            // unused; B frags use ab (same K order as A)

  // prologue: stage 0 into buffer 0
  stage16(gA,            &As[0][ldr * LDA + ldc]);
  stage16(gB0,           &Bs[0][ldr * LDA + ldc]);
  stage16(gB1,           &Bs[0][(ldr + 64) * LDA + ldc]);

#pragma unroll 1
  for (int k = 0; k < NK; ++k) {
    const int cur = k & 1;
    const int kk1 = (k + 1) * 32;
    if (k + 1 < NK) {                              // stage k+1 into other buffer
      stage16(gA  + kk1, &As[cur ^ 1][ldr * LDA + ldc]);
      stage16(gB0 + kk1, &Bs[cur ^ 1][ldr * LDA + ldc]);
      stage16(gB1 + kk1, &Bs[cur ^ 1][(ldr + 64) * LDA + ldc]);
      if (k + 2 < NK) {                            // global_prefetch_b8 of slab k+2
        __builtin_prefetch(gA + kk1 + 32, 0, 1);
        __builtin_prefetch(gB0 + kk1 + 32, 0, 1);
      }
      stage_wait_3();                              // my stage-k asyncs done (in-order)
    } else {
      stage_wait_0();
    }
    __syncthreads();                               // all stage-k data visible in LDS

    const unsigned short* Ac = As[cur];
    const unsigned short* Bc = Bs[cur];

    // --- build A fragments: lane m = l15; VGPR j<4 -> K=ab+2j, j>=4 -> K=ab+16+2(j-4)
    v16bf afrag[2], bfrag[2];
#pragma unroll
    for (int mi = 0; mi < 2; ++mi) {
      const int mloc = wm * 32 + mi * 16 + l15;
#pragma unroll
      for (int j = 0; j < 8; ++j) {
        const int k2 = ab + ((j < 4) ? (2 * j) : (16 + 2 * (j - 4)));
        PairCvt c; c.u = *(const unsigned*)&Ac[mloc * LDA + k2];
        afrag[mi][2 * j]     = c.h[0];
        afrag[mi][2 * j + 1] = c.h[1];
      }
    }
    // --- build B fragments: lane n = l15; VGPR j<4 -> K=ab+2j, j>=4 -> K=ab+16+2(j-4)
#pragma unroll
    for (int ni = 0; ni < 2; ++ni) {
      const int nloc = wn * 32 + ni * 16 + l15;
#pragma unroll
      for (int j = 0; j < 8; ++j) {
        PairCvt c; c.u = *(const unsigned*)&Bc[nloc * LDA + ab + ((j < 4) ? (2 * j) : (16 + 2 * (j - 4)))];
        bfrag[ni][2 * j]     = c.h[0];
        bfrag[ni][2 * j + 1] = c.h[1];
      }
    }

    // --- 4 WMMAs: D = A x B + C  (f32 accumulate) ---
    acc[0][0] = __builtin_amdgcn_wmma_f32_16x16x32_bf16(false, afrag[0], false, bfrag[0], (short)0, acc[0][0], false, false);
    acc[0][1] = __builtin_amdgcn_wmma_f32_16x16x32_bf16(false, afrag[0], false, bfrag[1], (short)0, acc[0][1], false, false);
    acc[1][0] = __builtin_amdgcn_wmma_f32_16x16x32_bf16(false, afrag[1], false, bfrag[0], (short)0, acc[1][0], false, false);
    acc[1][1] = __builtin_amdgcn_wmma_f32_16x16x32_bf16(false, afrag[1], false, bfrag[1], (short)0, acc[1][1], false, false);

    __syncthreads();                               // buf[cur] free for reuse at k+2
  }

  // --- epilogue: bias add + store (C/D layout: VGPR v -> M = v + (lane>=16 ? 8:0))
  const int moff = (lane & 16) ? 8 : 0;
#pragma unroll
  for (int ni = 0; ni < 2; ++ni) {
    const int n = n0 + wn * 32 + ni * 16 + l15;
    const float bias = b_enc[n];
#pragma unroll
    for (int mi = 0; mi < 2; ++mi) {
      const int mbase = m0 + wm * 32 + mi * 16 + moff;
#pragma unroll
      for (int v = 0; v < 8; ++v)
        z[(size_t)(mbase + v) * D_HID + n] = acc[mi][ni][v] + bias;
    }
  }
}

// ---------------------------------------------------------------------------
// Kernel 3: per-row top-K via binary search on order-preserving uint keys,
//   in-place masking of z -> z_sparse, deterministic compaction of (val,idx).
// ---------------------------------------------------------------------------
__global__ __launch_bounds__(256)
void topk_kernel(float* __restrict__ z, float* __restrict__ tv, int* __restrict__ ti) {
  const int row = blockIdx.x;
  const int tid = threadIdx.x;
  float* zr = z + (size_t)row * D_HID;

  __shared__ unsigned red[256];
  __shared__ unsigned offs[256];

  unsigned lo = 0u, hi = 0xFFFFFFFFu;
#pragma unroll 1
  for (int it = 0; it < 32; ++it) {
    const unsigned span = hi - lo;
    const unsigned mid  = lo + (span >> 1) + (span & 1u);   // ceil midpoint
    unsigned c = 0;
#pragma unroll 4
    for (int i = 0; i < D_HID / 256; ++i) {
      const unsigned u   = __float_as_uint(zr[tid + (i << 8)]);
      const unsigned key = (u & 0x80000000u) ? ~u : (u | 0x80000000u);
      c += (key >= mid) ? 1u : 0u;
    }
    red[tid] = c;
    __syncthreads();
    for (int s = 128; s > 0; s >>= 1) {
      if (tid < s) red[tid] += red[tid + s];
      __syncthreads();
    }
    const unsigned total = red[0];
    __syncthreads();
    if (total >= (unsigned)KK) lo = mid; else hi = mid - 1u;
    if (lo >= hi) break;
  }
  const unsigned T = lo;   // largest key-threshold with count(key>=T) >= K

  // per-thread keep counts -> exclusive scan (deterministic order)
  unsigned keep = 0;
#pragma unroll 4
  for (int i = 0; i < D_HID / 256; ++i) {
    const unsigned u   = __float_as_uint(zr[tid + (i << 8)]);
    const unsigned key = (u & 0x80000000u) ? ~u : (u | 0x80000000u);
    keep += (key >= T) ? 1u : 0u;
  }
  red[tid] = keep;
  __syncthreads();
  if (tid == 0) {
    unsigned run = 0;
    for (int t = 0; t < 256; ++t) { offs[t] = run; run += red[t]; }
  }
  __syncthreads();

  unsigned p = offs[tid];
  float* tvr = tv + (size_t)row * KK;
  int*   tir = ti + (size_t)row * KK;
#pragma unroll 4
  for (int i = 0; i < D_HID / 256; ++i) {
    const int h = tid + (i << 8);
    const float v = zr[h];
    const unsigned u   = __float_as_uint(v);
    const unsigned key = (u & 0x80000000u) ? ~u : (u | 0x80000000u);
    if (key >= T) {
      if (p < (unsigned)KK) { tvr[p] = v; tir[p] = h; }
      ++p;
    } else {
      zr[h] = 0.0f;                    // mask out -> z_sparse in place
    }
  }
}

// ---------------------------------------------------------------------------
// Kernel 4: sparse decode  x_hat[b,:] = b_dec + sum_j tv[j] * W_decT[ti[j], :]
// ---------------------------------------------------------------------------
__global__ __launch_bounds__(256)
void decode_kernel(const float* __restrict__ tv, const int* __restrict__ ti,
                   const float* __restrict__ wdT, const float* __restrict__ b_dec,
                   float* __restrict__ xhat) {
  const int row = blockIdx.x;
  const int tid = threadIdx.x;
  __shared__ float sv[KK];
  __shared__ int   si[KK];
  if (tid < KK) { sv[tid] = tv[(size_t)row * KK + tid]; si[tid] = ti[(size_t)row * KK + tid]; }
  __syncthreads();

  float a0 = b_dec[tid], a1 = b_dec[tid + 256], a2 = b_dec[tid + 512];
#pragma unroll 4
  for (int j = 0; j < KK; ++j) {
    const float v = sv[j];
    const float* w = wdT + (size_t)si[j] * D_IN;
    a0 += v * w[tid];
    a1 += v * w[tid + 256];
    a2 += v * w[tid + 512];
  }
  float* o = xhat + (size_t)row * D_IN;
  o[tid] = a0; o[tid + 256] = a1; o[tid + 512] = a2;
}

// ---------------------------------------------------------------------------
// Launch
// ---------------------------------------------------------------------------
extern "C" void kernel_launch(void* const* d_in, const int* in_sizes, int n_in,
                              void* d_out, int out_size, void* d_ws, size_t ws_size,
                              hipStream_t stream) {
  const float* x     = (const float*)d_in[0];
  const float* W_enc = (const float*)d_in[1];
  const float* b_enc = (const float*)d_in[2];
  const float* W_dec = (const float*)d_in[3];
  const float* b_dec = (const float*)d_in[4];
  (void)in_sizes; (void)n_in; (void)out_size;

  float* z    = (float*)d_out;                            // z_sparse [B, D_HID]
  float* xhat = (float*)d_out + (size_t)BB * D_HID;       // x_hat    [B, D_IN]

  // workspace layout (bytes)
  char* ws = (char*)d_ws;
  const size_t xb_bytes  = (size_t)BB * D_IN * 2;         // 12.6 MB
  const size_t wb_bytes  = (size_t)D_HID * D_IN * 2;      // 25.2 MB
  const size_t wdT_bytes = (size_t)D_HID * D_IN * 4;      // 50.3 MB
  const size_t tv_bytes  = (size_t)BB * KK * 4;           //  2.1 MB
  __bf16* xb  = (__bf16*)ws;
  __bf16* wb  = (__bf16*)(ws + xb_bytes);
  float*  wdT = (float*)(ws + xb_bytes + wb_bytes);
  float*  tv  = (float*)(ws + xb_bytes + wb_bytes + wdT_bytes);
  int*    ti  = (int*)(ws + xb_bytes + wb_bytes + wdT_bytes + tv_bytes);
  (void)ws_size;

  cvt_kernel<<<4096, 256, 0, stream>>>(x, W_enc, xb, wb);
  transpose_kernel<<<dim3(D_HID / 32, D_IN / 32), dim3(32, 8), 0, stream>>>(W_dec, wdT);
  encode_kernel<<<dim3(D_HID / 128, BB / 64), 256, 0, stream>>>(xb, wb, b_enc, z);
  topk_kernel<<<BB, 256, 0, stream>>>(z, tv, ti);
  decode_kernel<<<BB, 256, 0, stream>>>(tv, ti, wdT, b_dec, xhat);
}
